// WeightedMultiLabelTripletLoss_51788715655787
// MI455X (gfx1250) — 16-bit and 8-bit matrix operand order repaired and checked by reading, compiled, not run on hardware
//
#include <hip/hip_runtime.h>
#include <hip/hip_bf16.h>

#define B_SZ   8192
#define D_SZ   512
#define CPAD   128
#define MARGIN 0.3f
#define BIGF   1e9f

typedef __bf16 v16bf __attribute__((ext_vector_type(16)));
typedef __bf16 v8bf  __attribute__((ext_vector_type(8)));
typedef float  v8f   __attribute__((ext_vector_type(8)));
typedef int    v8i   __attribute__((ext_vector_type(8)));

__device__ __forceinline__ unsigned lds_off(const void* p) {
    return (unsigned)(unsigned long long)p;
}


__device__ __forceinline__ v16bf load_a_lds(const __bf16* smA, int k0, int lane) {
    const int m  = lane & 15;
    const int hi = lane >> 4;
    const __bf16* p = smA + m * D_SZ + k0 + 8 * hi;
    v8bf lo  = *(const v8bf*)(p);
    v8bf hi8 = *(const v8bf*)(p + 16);
    v16bf a;
#pragma unroll
    for (int t = 0; t < 8; ++t) { a[t] = lo[t]; a[t + 8] = hi8[t]; }
    return a;
}

__device__ __forceinline__ v16bf load_b_bf16(const __bf16* __restrict__ e,
                                             int col_base, int k0, int lane) {
    const int n  = lane & 15;
    const int hi = lane >> 4;
    const __bf16* p = e + (size_t)(col_base + n) * D_SZ + k0 + 8 * hi;
    v8bf lo  = *(const v8bf*)(p);
    v8bf hi8 = *(const v8bf*)(p + 16);
    v16bf b;
#pragma unroll
    for (int t = 0; t < 8; ++t) { b[t] = lo[t]; b[t + 8] = hi8[t]; }
    return b;
}

__device__ __forceinline__ v8i load_a_u8_lds(const unsigned char* smL, int k0, int lane) {
    const int m  = lane & 15;
    const int hi = lane >> 4;
    const unsigned char* p = smL + m * CPAD + k0 + 16 * hi;
    uint2 c0 = *(const uint2*)(p);
    uint2 c1 = *(const uint2*)(p + 8);
    uint2 c2 = *(const uint2*)(p + 32);
    uint2 c3 = *(const uint2*)(p + 40);
    v8i a;
    a[0] = (int)c0.x; a[1] = (int)c0.y; a[2] = (int)c1.x; a[3] = (int)c1.y;
    a[4] = (int)c2.x; a[5] = (int)c2.y; a[6] = (int)c3.x; a[7] = (int)c3.y;
    return a;
}

__device__ __forceinline__ v8i load_b_u8(const unsigned char* __restrict__ lab,
                                         int col_base, int k0, int lane) {
    const int n  = lane & 15;
    const int hi = lane >> 4;
    const unsigned char* p = lab + (size_t)(col_base + n) * CPAD + k0 + 16 * hi;
    uint4 c0 = *(const uint4*)(p);
    uint4 c1 = *(const uint4*)(p + 32);
    v8i b;
    b[0] = (int)c0.x; b[1] = (int)c0.y; b[2] = (int)c0.z; b[3] = (int)c0.w;
    b[4] = (int)c1.x; b[5] = (int)c1.y; b[6] = (int)c1.z; b[7] = (int)c1.w;
    return b;
}


__global__ void prep_embed(const float* __restrict__ emb,
                           __bf16* __restrict__ e, float* __restrict__ sq) {
    const int row  = blockIdx.x * 8 + (threadIdx.x >> 5);
    const int lane = threadIdx.x & 31;
    const float* r = emb + (size_t)row * D_SZ;
    float vals[16];
    float ss = 0.f;
#pragma unroll
    for (int t = 0; t < 16; ++t) { float x = r[lane + 32 * t]; vals[t] = x; ss += x * x; }
#pragma unroll
    for (int m = 1; m < 32; m <<= 1) ss += __shfl_xor(ss, m, 32);
    const float inv = 1.f / fmaxf(sqrtf(ss), 1e-12f);
    __bf16* o = e + (size_t)row * D_SZ;
    float ss2 = 0.f;
#pragma unroll
    for (int t = 0; t < 16; ++t) {
        __bf16 bv = (__bf16)(vals[t] * inv);
        o[lane + 32 * t] = bv;
        float f = (float)bv;
        ss2 += f * f;
    }
#pragma unroll
    for (int m = 1; m < 32; m <<= 1) ss2 += __shfl_xor(ss2, m, 32);
    if (lane == 0) sq[row] = ss2;
}

__global__ void prep_labels(const float* __restrict__ labels,
                            unsigned char* __restrict__ lab, float* __restrict__ sv) {
    const int row  = blockIdx.x * 8 + (threadIdx.x >> 5);
    const int lane = threadIdx.x & 31;
    const float* r = labels + (size_t)row * 80;
    unsigned char* o = lab + (size_t)row * CPAD;
    float s = 0.f;
#pragma unroll
    for (int t = 0; t < 4; ++t) {
        int c = lane + 32 * t;
        float x = (c < 80) ? r[c] : 0.f;
        unsigned char v = (x != 0.f) ? 1 : 0;
        o[c] = v;
        s += (float)v;
    }
#pragma unroll
    for (int m = 1; m < 32; m <<= 1) s += __shfl_xor(s, m, 32);
    if (lane == 0) sv[row] = s;
}

__global__ void init_acc(unsigned int* __restrict__ hn,
                         float* __restrict__ rh, unsigned int* __restrict__ rp) {
    int i = blockIdx.x * blockDim.x + threadIdx.x;
    if (i < B_SZ) { hn[i] = __float_as_uint(BIGF); rh[i] = 0.f; rp[i] = 0u; }
}

template <bool LOSS>
__global__ void gram_pass(const __bf16* __restrict__ e,
                          const unsigned char* __restrict__ lab,
                          const float* __restrict__ sq,
                          const float* __restrict__ sv,
                          unsigned int* __restrict__ hn_bits,
                          float* __restrict__ row_hinge,
                          unsigned int* __restrict__ row_pos) {
    __shared__ __bf16        smA[16 * D_SZ];
    __shared__ unsigned char smL[16 * CPAD];

    const int tid   = threadIdx.x;
    const int lane  = tid & 31;
    const int wave  = tid >> 5;
    const int i0    = blockIdx.x * 16;
    const int jbase = blockIdx.y * 512 + wave * 64;

    {
        const char* gA = (const char*)(e + (size_t)i0 * D_SZ) + tid * 64;
        unsigned    dA = lds_off(smA) + (unsigned)tid * 64u;
#pragma unroll
        for (int c = 0; c < 4; ++c) {
            asm volatile("global_load_async_to_lds_b128 %0, %1, off"
                         :: "v"(dA + (unsigned)c * 16u), "v"(gA + c * 16) : "memory");
        }
        const char* gL = (const char*)(lab + (size_t)i0 * CPAD) + tid * 8;
        unsigned    dL = lds_off(smL) + (unsigned)tid * 8u;
        asm volatile("global_load_async_to_lds_b64 %0, %1, off"
                     :: "v"(dL), "v"(gL) : "memory");
        asm volatile("s_wait_asynccnt 0x0" ::: "memory");
    }
    __syncthreads();

    v8f facc[4];
#pragma unroll
    for (int t = 0; t < 4; ++t) facc[t] = (v8f){0.f,0.f,0.f,0.f,0.f,0.f,0.f,0.f};

    for (int k0 = 0; k0 < D_SZ; k0 += 32) {
        v16bf a  = load_a_lds(smA, k0, lane);
        v16bf b0 = load_b_bf16(e, jbase +  0, k0, lane);
        v16bf b1 = load_b_bf16(e, jbase + 16, k0, lane);
        v16bf b2 = load_b_bf16(e, jbase + 32, k0, lane);
        v16bf b3 = load_b_bf16(e, jbase + 48, k0, lane);
        facc[0] = __builtin_amdgcn_wmma_f32_16x16x32_bf16(false, a, false, b0, (short)0, facc[0], false, false);
        facc[1] = __builtin_amdgcn_wmma_f32_16x16x32_bf16(false, a, false, b1, (short)0, facc[1], false, false);
        facc[2] = __builtin_amdgcn_wmma_f32_16x16x32_bf16(false, a, false, b2, (short)0, facc[2], false, false);
        facc[3] = __builtin_amdgcn_wmma_f32_16x16x32_bf16(false, a, false, b3, (short)0, facc[3], false, false);
    }

    uint2 pk[4];
#pragma unroll
    for (int t = 0; t < 4; ++t) {
        v8i ia = (v8i){0,0,0,0,0,0,0,0};
#pragma unroll
        for (int k0 = 0; k0 < CPAD; k0 += 64) {
            v8i a8 = load_a_u8_lds(smL, k0, lane);
            v8i b8 = load_b_u8(lab, jbase + t * 16, k0, lane);
            ia = __builtin_amdgcn_wmma_i32_16x16x64_iu8(false, a8, false, b8, ia, false, false);
        }
        pk[t].x = (unsigned)ia[0] | ((unsigned)ia[1] << 8) | ((unsigned)ia[2] << 16) | ((unsigned)ia[3] << 24);
        pk[t].y = (unsigned)ia[4] | ((unsigned)ia[5] << 8) | ((unsigned)ia[6] << 16) | ((unsigned)ia[7] << 24);
    }

    const int hi = lane >> 4, n = lane & 15;
    float sqi[8];
#pragma unroll
    for (int v = 0; v < 8; ++v) sqi[v] = sq[i0 + v + 8 * hi];

    if (!LOSS) {
        float minv[8];
#pragma unroll
        for (int v = 0; v < 8; ++v) minv[v] = BIGF;
#pragma unroll
        for (int t = 0; t < 4; ++t) {
            float sqj = sq[jbase + t * 16 + n];
#pragma unroll
            for (int v = 0; v < 8; ++v) {
                unsigned w  = (v < 4) ? pk[t].x : pk[t].y;
                unsigned iv = (w >> (8 * (v & 3))) & 0xffu;
                float d2   = sqi[v] + sqj - 2.f * facc[t][v];
                float dist = sqrtf(fmaxf(d2, 0.f));
                if (iv == 0u) minv[v] = fminf(minv[v], dist);
            }
        }
#pragma unroll
        for (int m = 1; m < 16; m <<= 1)
#pragma unroll
            for (int v = 0; v < 8; ++v)
                minv[v] = fminf(minv[v], __shfl_xor(minv[v], m, 32));
        if (n == 0) {
#pragma unroll
            for (int v = 0; v < 8; ++v)
                atomicMin(&hn_bits[i0 + v + 8 * hi], __float_as_uint(minv[v]));
        }
    } else {
        float si[8], hnv[8], hsum[8];
        unsigned int pcnt[8];
#pragma unroll
        for (int v = 0; v < 8; ++v) {
            int ir = i0 + v + 8 * hi;
            si[v]   = sv[ir];
            hnv[v]  = __uint_as_float(hn_bits[ir]);
            hsum[v] = 0.f; pcnt[v] = 0u;
        }
#pragma unroll
        for (int t = 0; t < 4; ++t) {
            int jr = jbase + t * 16 + n;
            float sqj = sq[jr];
            float sj  = sv[jr];
#pragma unroll
            for (int v = 0; v < 8; ++v) {
                int ir = i0 + v + 8 * hi;
                unsigned w  = (v < 4) ? pk[t].x : pk[t].y;
                unsigned iv = (w >> (8 * (v & 3))) & 0xffu;
                if (iv > 0u && ir != jr) {
                    float d2    = sqi[v] + sqj - 2.f * facc[t][v];
                    float dist  = sqrtf(fmaxf(d2, 0.f));
                    float inter = (float)iv;
                    float jac   = inter / (si[v] + sj - inter + 1e-8f);
                    float hinge = fmaxf(dist - hnv[v] + MARGIN, 0.f) * jac;
                    hsum[v] += hinge;
                    pcnt[v] += 1u;
                }
            }
        }
#pragma unroll
        for (int m = 1; m < 16; m <<= 1)
#pragma unroll
            for (int v = 0; v < 8; ++v) {
                hsum[v] += __shfl_xor(hsum[v], m, 32);
                pcnt[v] += (unsigned int)__shfl_xor((int)pcnt[v], m, 32);
            }
        if (n == 0) {
#pragma unroll
            for (int v = 0; v < 8; ++v) {
                int ir = i0 + v + 8 * hi;
                atomicAdd(&row_hinge[ir], hsum[v]);
                atomicAdd(&row_pos[ir], pcnt[v]);
            }
        }
    }
}


__global__ void finalize(const unsigned int* __restrict__ hn_bits,
                         const float* __restrict__ rh,
                         const unsigned int* __restrict__ rp,
                         float* __restrict__ out) {
    __shared__ float s_sum[256];
    __shared__ float s_cnt[256];
    const int tid = threadIdx.x;
    float sum = 0.f, cnt = 0.f;
    for (int i = tid; i < B_SZ; i += 256) {
        unsigned int pc = rp[i];
        float hn = __uint_as_float(hn_bits[i]);
        bool valid = (pc > 0u) && (hn < 0.5f * BIGF);
        float rl = rh[i] / (float)(pc > 0u ? pc : 1u);
        if (valid) { sum += rl; cnt += 1.f; }
    }
    s_sum[tid] = sum; s_cnt[tid] = cnt;
    __syncthreads();
    for (int st = 128; st > 0; st >>= 1) {
        if (tid < st) { s_sum[tid] += s_sum[tid + st]; s_cnt[tid] += s_cnt[tid + st]; }
        __syncthreads();
    }
    if (tid == 0) { out[0] = s_sum[0] / (s_cnt[0] + 1e-8f); out[1] = 0.f; }
}


extern "C" void kernel_launch(void* const* d_in, const int* in_sizes, int n_in,
                              void* d_out, int out_size, void* d_ws, size_t ws_size,
                              hipStream_t stream) {
    const float* emb    = (const float*)d_in[0];
    const float* labels = (const float*)d_in[1];
    float* out = (float*)d_out;

    char* ws = (char*)d_ws;
    size_t off = 0;
    __bf16*        e   = (__bf16*)(ws + off);        off += (size_t)B_SZ * D_SZ * 2;
    unsigned char* lab = (unsigned char*)(ws + off); off += (size_t)B_SZ * CPAD;
    float*         sq  = (float*)(ws + off);         off += (size_t)B_SZ * 4;
    float*         sv  = (float*)(ws + off);         off += (size_t)B_SZ * 4;
    unsigned int*  hn  = (unsigned int*)(ws + off);  off += (size_t)B_SZ * 4;
    float*         rh  = (float*)(ws + off);         off += (size_t)B_SZ * 4;
    unsigned int*  rp  = (unsigned int*)(ws + off);  off += (size_t)B_SZ * 4;

    prep_embed <<<B_SZ / 8, 256, 0, stream>>>(emb, e, sq);
    prep_labels<<<B_SZ / 8, 256, 0, stream>>>(labels, lab, sv);
    init_acc   <<<B_SZ / 256, 256, 0, stream>>>(hn, rh, rp);

    dim3 grid(B_SZ / 16, 16);
    gram_pass<false><<<grid, 256, 0, stream>>>(e, lab, sq, sv, hn, rh, rp);
    gram_pass<true> <<<grid, 256, 0, stream>>>(e, lab, sq, sv, hn, rh, rp);

    finalize<<<1, 256, 0, stream>>>(hn, rh, rp, out);
}
